// channel_attention_77515569758592
// MI455X (gfx1250) — compile-verified, matrix operand order checked by reading, not run on hardware
//
#include <hip/hip_runtime.h>

typedef __attribute__((ext_vector_type(16))) __bf16 v16bf;
typedef __attribute__((ext_vector_type(8)))  __bf16 v8bf;
typedef __attribute__((ext_vector_type(8)))  float  v8f;

constexpr int NC    = 128;   // channels
constexpr int NS    = 400;   // sequence
constexpr int NST   = 26;    // s-tiles: 25 over s<400 + 1 zero pad tile
constexpr int NSPAD = 416;   // padded rows
constexpr int TSTR  = 136;   // row stride (bf16 elems): 272B = 17*16 -> 16B aligned rows
constexpr int INTER = 10;
constexpr int NSP   = 40;    // pooled length
constexpr int SPK   = 64;    // K padded for attention GEMM
constexpr int QPSTR = 48;    // fp32 pool accumulator stride
constexpr float EPS = 1e-5f;

// ---- LDS layout (bytes) -------------------------------------------------
constexpr int OFF_TEMP = 0;                           // bf16 [416][136]  (temp, later A2)
constexpr int SZ_TEMP  = NSPAD * TSTR * 2;            // 113152
constexpr int OFF_W    = OFF_TEMP + SZ_TEMP;          // bf16 [2][128][136] (Wq/Wk, later Wp)
constexpr int SZ_W1    = NC * TSTR * 2;               // 34816
constexpr int OFF_SCR  = OFF_W + 2 * SZ_W1;           // 182784: time-multiplexed scratch
constexpr int SZ_SCR   = NSPAD * TSTR * 2;            // 113152 (p-buffer bf16 [416][136])
constexpr int OFF_QPB  = OFF_SCR;                     // bf16 [2][128][64]  = 32768
constexpr int OFF_QPF  = OFF_SCR + 2 * NC * SPK * 2;  // f32  [2][128][48]  = 49152
constexpr int OFF_ATT  = OFF_QPF;                     // f32  [8][16][128]  = 65536 (aliases QPF, sequenced)
constexpr int OFF_R    = OFF_QPF + 8 * 16 * NC * 4;   // f32  [128]
constexpr int OFF_PAR  = OFF_SCR + SZ_SCR;            // f32  [1408] params
constexpr int SMEM_BYTES = OFF_PAR + 1408 * 4;        // 301568 <= 320K WGP LDS

// ---- helpers ------------------------------------------------------------
__device__ __forceinline__ unsigned short f2bf(float f) {
  unsigned u = __float_as_uint(f);
  u += 0x7FFFu + ((u >> 16) & 1u);        // round-to-nearest-even
  return (unsigned short)(u >> 16);
}
__device__ __forceinline__ float bf2f(unsigned short h) {
  return __uint_as_float(((unsigned)h) << 16);
}
__device__ __forceinline__ void frag_load(v16bf* f, const unsigned short* base,
                                          int row, int rowStride, int koff) {
  // 16-bit A/B fragment: lanes 0-15 row M, K {k0..k0+7, k0+16..k0+23};
  // lanes 16-31 same rows, K +8 / +24. Two aligned 16B LDS reads.
  const unsigned short* p = base + row * rowStride + koff;
  union { v16bf v; v8bf h[2]; } u;
  u.h[0] = *(const v8bf*)(p);
  u.h[1] = *(const v8bf*)(p + 16);
  *f = u.v;
}
__device__ __forceinline__ v8f wmma_bf16(v16bf a, v16bf b, v8f c) {
  return __builtin_amdgcn_wmma_f32_16x16x32_bf16(false, a, false, b, (short)0, c,
                                                 false, false);
}

// Row-block GEMM [16 x 128] = A[16 x 128] * W^T, + bias, per-row LayerNorm.
// MODE 0: pool (avg over 10 along s) into qpDst via ds_add_f32.
// MODE 1: write LN'd result bf16 to pDst[s][n].
template <int MODE>
__device__ __forceinline__ void gemm_ln(const unsigned short* A, const unsigned short* W,
                                        const float* pb /* [bias|g|beta] x128 */,
                                        float* qpDst, unsigned short* pDst) {
  const int lane = threadIdx.x & 31;
  const int lm   = lane & 15;
  const int half = lane >> 4;
  const int wave = threadIdx.x >> 5;
  for (int ti = wave; ti < NST; ti += 8) {
    v16bf afr[4];
#pragma unroll
    for (int kk = 0; kk < 4; kk++)
      frag_load(&afr[kk], A, ti * 16 + lm, TSTR, kk * 32 + half * 8);
    v8f acc[8];
#pragma unroll
    for (int t = 0; t < 8; t++) {
      v8f c = {0.f, 0.f, 0.f, 0.f, 0.f, 0.f, 0.f, 0.f};
#pragma unroll
      for (int kk = 0; kk < 4; kk++) {
        v16bf bfr;
        frag_load(&bfr, W, t * 16 + lm, TSTR, kk * 32 + half * 8);
        c = wmma_bf16(afr[kk], bfr, c);
      }
      acc[t] = c;
    }
    // bias + per-row (per s) stats; row m = r + 8*half lives in this half's lanes
    float sums[8], sqs[8];
#pragma unroll
    for (int r = 0; r < 8; r++) { sums[r] = 0.f; sqs[r] = 0.f; }
#pragma unroll
    for (int t = 0; t < 8; t++) {
      float bia = pb[t * 16 + lm];
#pragma unroll
      for (int r = 0; r < 8; r++) {
        float v = acc[t][r] + bia;
        acc[t][r] = v;
        sums[r] += v; sqs[r] += v * v;
      }
    }
#pragma unroll
    for (int r = 0; r < 8; r++) {
#pragma unroll
      for (int m = 1; m < 16; m <<= 1) {   // reduce within 16-lane half (wave32)
        sums[r] += __shfl_xor(sums[r], m, 32);
        sqs[r]  += __shfl_xor(sqs[r], m, 32);
      }
    }
#pragma unroll
    for (int t = 0; t < 8; t++) {
      int n = t * 16 + lm;
      float g = pb[128 + n], be = pb[256 + n];
#pragma unroll
      for (int r = 0; r < 8; r++) {
        float mean = sums[r] * (1.f / NC);
        float var  = sqs[r] * (1.f / NC) - mean * mean;
        float rstd = rsqrtf(var + EPS);
        float y = (acc[t][r] - mean) * rstd * g + be;
        int s = ti * 16 + r + half * 8;
        if (s < NS) {
          if (MODE == 0)
            atomicAdd(&qpDst[n * QPSTR + s / INTER], y * (1.f / INTER));
          else
            pDst[s * TSTR + n] = f2bf(y);
        }
      }
    }
  }
}

__global__ __launch_bounds__(256, 1)
void chanattn_kernel(const float* __restrict__ x,
                     const float* ln0_g, const float* ln0_b,
                     const float* Wq, const float* bq, const float* gq, const float* betq,
                     const float* Wk, const float* bk, const float* gk, const float* betk,
                     const float* Wp, const float* bp, const float* gp, const float* betp,
                     float* __restrict__ out) {
  extern __shared__ char smem[];
  unsigned short* tempU = (unsigned short*)(smem + OFF_TEMP);
  unsigned short* wU    = (unsigned short*)(smem + OFF_W);
  unsigned short* qpbU  = (unsigned short*)(smem + OFF_QPB);
  float*          qpF   = (float*)(smem + OFF_QPF);
  float*          attS  = (float*)(smem + OFF_ATT);
  float*          rS    = (float*)(smem + OFF_R);
  float*          par   = (float*)(smem + OFF_PAR);
  unsigned short* pbU   = (unsigned short*)(smem + OFF_SCR);

  const int tid  = threadIdx.x;
  const int lane = tid & 31;
  const int wave = tid >> 5;
  const float* xs   = x   + (size_t)blockIdx.x * NC * NS;
  float*       outs = out + (size_t)blockIdx.x * NC * NS;

  __builtin_prefetch(xs + tid * 32, 0, 0);   // warm L2 (global_prefetch_b8)

  // ---- (a) params, weights Wq/Wk -> bf16 LDS, zero pool accum & pad rows
  if (tid < 128) {
    int i = tid;
    par[i] = ln0_g[i];        par[128 + i] = ln0_b[i];
    par[256 + i]  = bq[i];    par[384 + i]  = gq[i];   par[512 + i]  = betq[i];
    par[640 + i]  = bk[i];    par[768 + i]  = gk[i];   par[896 + i]  = betk[i];
    par[1024 + i] = bp[i];    par[1152 + i] = gp[i];   par[1280 + i] = betp[i];
  }
  for (int i = tid; i < NC * NC; i += 256) {
    int n = i >> 7, c0 = i & 127;
    wU[n * TSTR + c0]              = f2bf(Wq[i]);
    wU[NC * TSTR + n * TSTR + c0]  = f2bf(Wk[i]);
  }
  for (int i = tid; i < 2 * NC * QPSTR; i += 256) qpF[i] = 0.f;
  for (int i = tid; i < 16 * TSTR; i += 256) tempU[NS * TSTR + i] = 0;  // pad rows
  __syncthreads();

  // ---- (b) temp = LN_c(x^T), bf16 (coalesced: consecutive tids = consecutive s)
  for (int s = tid; s < NS; s += 256) {
    float sum = 0.f, sq = 0.f;
    for (int c0 = 0; c0 < NC; c0++) {
      float v = xs[c0 * NS + s];
      sum += v; sq += v * v;
    }
    float mean = sum * (1.f / NC);
    float rstd = rsqrtf(sq * (1.f / NC) - mean * mean + EPS);
    for (int c0 = 0; c0 < NC; c0++) {
      float v = xs[c0 * NS + s];
      tempU[s * TSTR + c0] = f2bf((v - mean) * rstd * par[c0] + par[128 + c0]);
    }
  }
  __syncthreads();

  // ---- (c) q/k GEMM + LN + pooled accumulate (ds_add_f32)
  gemm_ln<0>(tempU, wU,            par + 256, qpF,              nullptr);  // q
  gemm_ln<0>(tempU, wU + NC * TSTR, par + 640, qpF + NC * QPSTR, nullptr);  // k
  __syncthreads();

  // ---- (d) pooled fp32 -> bf16 [2][128][64] (K zero-padded 40..63)
  for (int i = tid; i < 2 * NC * SPK; i += 256) {
    int mat = i / (NC * SPK), rem = i % (NC * SPK);
    int c0 = rem >> 6, sp = rem & 63;
    float v = (sp < NSP) ? qpF[mat * NC * QPSTR + c0 * QPSTR + sp] : 0.f;
    qpbU[i] = f2bf(v);
  }
  __syncthreads();

  // ---- (e) atten = qp kp^T / sqrt(40); each wave owns a 16-row block
  {
    const int lm = lane & 15, half = lane >> 4;
    const unsigned short* kpB = qpbU + NC * SPK;
    v16bf afr0, afr1;
    frag_load(&afr0, qpbU, wave * 16 + lm, SPK, half * 8);
    frag_load(&afr1, qpbU, wave * 16 + lm, SPK, 32 + half * 8);
    const float sc = 0.15811388300841897f;  // 1/sqrt(40)
#pragma unroll
    for (int t = 0; t < 8; t++) {
      v8f c = {0.f, 0.f, 0.f, 0.f, 0.f, 0.f, 0.f, 0.f};
      v16bf bfr;
      frag_load(&bfr, kpB, t * 16 + lm, SPK, half * 8);
      c = wmma_bf16(afr0, bfr, c);
      frag_load(&bfr, kpB, t * 16 + lm, SPK, 32 + half * 8);
      c = wmma_bf16(afr1, bfr, c);
#pragma unroll
      for (int r = 0; r < 8; r++) {
        int m = r + half * 8;
        attS[wave * 2048 + m * 128 + t * 16 + lm] = c[r] * sc;
      }
    }
  }
  __syncthreads();
  // softmax rowsums -> r[c]  (faithfully sum score over m; == 1 up to fp rounding)
  for (int m = 0; m < 16; m++) {
    const float* row = attS + wave * 2048 + m * 128;
    float v0 = row[lane], v1 = row[lane + 32], v2 = row[lane + 64], v3 = row[lane + 96];
    float mx = fmaxf(fmaxf(v0, v1), fmaxf(v2, v3));
#pragma unroll
    for (int k = 1; k < 32; k <<= 1) mx = fmaxf(mx, __shfl_xor(mx, k, 32));
    float e0 = __expf(v0 - mx), e1 = __expf(v1 - mx), e2 = __expf(v2 - mx), e3 = __expf(v3 - mx);
    float ss = e0 + e1 + e2 + e3;
#pragma unroll
    for (int k = 1; k < 32; k <<= 1) ss += __shfl_xor(ss, k, 32);
    float q = e0 / ss + e1 / ss + e2 / ss + e3 / ss;
#pragma unroll
    for (int k = 1; k < 32; k <<= 1) q += __shfl_xor(q, k, 32);
    if (lane == 0) rS[wave * 16 + m] = q;
  }
  __syncthreads();

  // ---- (f) A2 = (x * r)^T bf16 into temp region; Wp into slot 0
  for (int s = tid; s < NS; s += 256)
    for (int c0 = 0; c0 < NC; c0++)
      tempU[s * TSTR + c0] = f2bf(xs[c0 * NS + s] * rS[c0]);
  for (int i = tid; i < NC * NC; i += 256) {
    int n = i >> 7, c0 = i & 127;
    wU[n * TSTR + c0] = f2bf(Wp[i]);
  }
  __syncthreads();

  // ---- (g) p = LN(A2 Wp^T + bp) -> bf16 p-buffer (scratch region reused)
  gemm_ln<1>(tempU, wU, par + 1024, nullptr, pbU);
  __syncthreads();

  // ---- (h) out[c][s] = p[s][c] + x[c][s], coalesced over s
  for (int idx = tid; idx < NC * NS; idx += 256) {
    int c0 = idx / NS, s = idx % NS;
    outs[idx] = bf2f(pbU[s * TSTR + c0]) + xs[idx];
  }
}

extern "C" void kernel_launch(void* const* d_in, const int* in_sizes, int n_in,
                              void* d_out, int out_size, void* d_ws, size_t ws_size,
                              hipStream_t stream) {
  (void)n_in; (void)out_size; (void)d_ws; (void)ws_size;
  const float* x     = (const float*)d_in[0];
  const float* ln0_g = (const float*)d_in[1];
  const float* ln0_b = (const float*)d_in[2];
  const float* Wq    = (const float*)d_in[3];
  const float* bq    = (const float*)d_in[4];
  const float* gq    = (const float*)d_in[5];
  const float* betq  = (const float*)d_in[6];
  const float* Wk    = (const float*)d_in[7];
  const float* bk    = (const float*)d_in[8];
  const float* gk    = (const float*)d_in[9];
  const float* betk  = (const float*)d_in[10];
  const float* Wp    = (const float*)d_in[11];
  const float* bp    = (const float*)d_in[12];
  const float* gp    = (const float*)d_in[13];
  const float* betp  = (const float*)d_in[14];
  float* out = (float*)d_out;

  int n_slices = in_sizes[0] / (NC * NS);  // 32*16 = 512
  hipFuncSetAttribute((const void*)chanattn_kernel,
                      hipFuncAttributeMaxDynamicSharedMemorySize, SMEM_BYTES);
  chanattn_kernel<<<n_slices, 256, SMEM_BYTES, stream>>>(
      x, ln0_g, ln0_b, Wq, bq, gq, betq, Wk, bk, gk, betk, Wp, bp, gp, betp, out);
}
